// ConvAttention_67783173865537
// MI455X (gfx1250) — matrix operand order repaired and checked by reading, not compiled, not run on hardware
//
#include <hip/hip_runtime.h>
#include <hip/hip_bf16.h>
#include <float.h>
#include <math.h>

typedef __attribute__((ext_vector_type(16))) _Float16 v16h;
typedef __attribute__((ext_vector_type(8)))  _Float16 v8h;
typedef __attribute__((ext_vector_type(4)))  _Float16 v4h;
typedef __attribute__((ext_vector_type(8)))  float    v8f;

#define BATCH  32
#define TT1    1600
#define TT2    400

#define LDS_PITCH 40   // halves per LDS row (80 B): bank-spread, 16B-aligned rows

__device__ __forceinline__ int iclamp(int v, int lo, int hi) {
    return v < lo ? lo : (v > hi ? hi : v);
}
__device__ __forceinline__ int imin(int a, int b) { return a < b ? a : b; }

// ---------------------------------------------------------------------------
// conv1d-as-GEMM with WMMA f16 (f32 accumulate); all dims compile-time.
//   Y[b, co, t] = act( bias[co] + sum_{tap, ci} W[co, ci, tap] * X[b, ci, t+tap-PAD] )
// Block: 128 threads (4 waves); block tile = 16(co) x 64(t), one 16x16 per wave.
// grid: (ceil(T/64), COUT/16, B)
// K-remainder handling: clamp base row-group to stay in bounds (loads always
// legal & unconditional), then zero invalid lanes with a branchless select.
// Requires CIN % 16 == 0 (true for all instantiations: 512,1024,80,160,80).
// ---------------------------------------------------------------------------
template <int CIN, int COUT, int T, int TAPS, int PAD, bool RELU>
__global__ __launch_bounds__(128)
void conv1d_wmma(const float* __restrict__ X, const float* __restrict__ W,
                 const float* __restrict__ bias, float* __restrict__ Y)
{
    constexpr bool GUARD_CI = (CIN % 32) != 0;
    static_assert(CIN % 16 == 0, "remainder scheme needs CIN % 16 == 0");

    __shared__ _Float16 As[16][LDS_PITCH];        // [co_r][k]
    __shared__ _Float16 Bs[4][16][LDS_PITCH];     // per wave: [t_n][k]

    const int tid  = threadIdx.x;
    const int wave = tid >> 5;
    const int lane = tid & 31;
    const int m    = lane & 15;
    const int hi   = lane >> 4;

    const int t0  = blockIdx.x * 64;
    const int co0 = blockIdx.y * 16;
    const int b   = blockIdx.z;

    const float* Xb = X + (size_t)b * CIN * T;

    // staging index assignments (per-lane contiguous along k -> packed stores)
    const int a_r  = tid >> 3;          // 0..15  (co row)
    const int a_c0 = (tid & 7) * 4;     // k base, 4 contiguous per thread
    const int b_n  = m;                 // t column within wave tile
    const int b_kh = hi;                // k half: 0 -> k 0..15, 1 -> k 16..31

    v8f acc = {};

    #pragma unroll
    for (int tap = 0; tap < TAPS; ++tap) {
        for (int ci0 = 0; ci0 < CIN; ci0 += 32) {
            __syncthreads();

            // ---- A tile (weights) 16x32: 4 contiguous k per thread ----
            {
                const int  ci_grp = GUARD_CI ? imin(ci0 + a_c0, CIN - 4) : (ci0 + a_c0);
                const bool aval   = !GUARD_CI || (ci0 + a_c0 + 3 < CIN);
                const float* wp = W + ((size_t)(co0 + a_r) * CIN + ci_grp) * TAPS + tap;
                v4h pk;
                #pragma unroll
                for (int it = 0; it < 4; ++it) {
                    float v = wp[it * TAPS];
                    if (GUARD_CI && !aval) v = 0.0f;
                    pk[it] = (_Float16)v;
                }
                *(v4h*)&As[a_r][a_c0] = pk;
            }

            // ---- B tile 32(k) x 16(n) per wave: 16 contiguous k per lane ----
            {
                const int  t    = t0 + wave * 16 + b_n + tap - PAD;
                const int  tc   = iclamp(t, 0, T - 1);
                const int  ci_grp = GUARD_CI ? imin(ci0 + b_kh * 16, CIN - 16)
                                             : (ci0 + b_kh * 16);
                const bool ok   = (t == tc) && (!GUARD_CI || (ci0 + b_kh * 16 < CIN));
                const float* xp = Xb + (size_t)ci_grp * T + tc;
                v8h p0, p1;
                #pragma unroll
                for (int it = 0; it < 16; ++it) {
                    float v = xp[it * T];
                    if (!ok) v = 0.0f;
                    if (it < 8) p0[it] = (_Float16)v;
                    else        p1[it - 8] = (_Float16)v;
                }
                *(v8h*)&Bs[wave][b_n][b_kh * 16]     = p0;
                *(v8h*)&Bs[wave][b_n][b_kh * 16 + 8] = p1;
            }
            __syncthreads();

            // ---- fragments: contiguous 16-bit runs -> ds_load_b128 ----
            v16h a, bf;
            #pragma unroll
            for (int i = 0; i < 16; ++i)
                a[i] = As[m][hi * 16 + i];
            #pragma unroll
            for (int i = 0; i < 16; ++i)
                bf[i] = Bs[wave][m][hi * 16 + i];

            acc = __builtin_amdgcn_wmma_f32_16x16x32_f16(
                false, a, false, bf, (short)0, acc, false, false);
        }
    }

    // output tile: base pointer + compile-time row-stride immediates
    {
        const int t = t0 + wave * 16 + m;
        const int co_base = co0 + 8 * hi;
        float* yp = Y + ((size_t)b * COUT + co_base) * T + t;
        #pragma unroll
        for (int r = 0; r < 8; ++r) {
            if (t < T) {
                float y = acc[r] + bias[co_base + r];
                if (RELU) y = fmaxf(y, 0.0f);
                yp[(size_t)r * T] = y;
            }
        }
    }
}

// ---------------------------------------------------------------------------
// qk[b, t, s] = sum_c q[b,c,t] * k[b,c,s]   (A = q^T; M=TT1, N=TT2, K=C=80)
// Block: 128 threads; waves split M: block = 64(t) x 16(s).
// grid: (TT2/16, TT1/64, B)
// ---------------------------------------------------------------------------
#define QK_C 80

__global__ __launch_bounds__(128)
void qk_wmma(const float* __restrict__ Q, const float* __restrict__ K,
             float* __restrict__ QK)
{
    __shared__ _Float16 As[4][16][LDS_PITCH];   // per wave: [t_r][c]
    __shared__ _Float16 Bs[16][LDS_PITCH];      // shared:   [s_n][c]

    const int tid  = threadIdx.x;
    const int wave = tid >> 5;
    const int lane = tid & 31;
    const int m    = lane & 15;
    const int hi   = lane >> 4;

    const int s0 = blockIdx.x * 16;
    const int t0 = blockIdx.y * 64;
    const int b  = blockIdx.z;

    const float* Qb = Q + (size_t)b * QK_C * TT1;
    const float* Kb = K + (size_t)b * QK_C * TT2;

    const int bn = tid & 15;            // s column
    const int bk = (tid >> 4) * 4;      // c base, 4 contiguous per thread
    const int ar = m;                   // t row within wave tile
    const int ah = hi;                  // c half

    v8f acc = {};

    #pragma unroll
    for (int c0 = 0; c0 < QK_C; c0 += 32) {
        __syncthreads();

        // B tile (k-matrix) 32(c) x 16(s): 4 contiguous c per thread
        {
            const int  c_grp = imin(c0 + bk, QK_C - 4);
            const bool ok    = (c0 + bk + 3 < QK_C);
            const float* kp = Kb + (size_t)c_grp * TT2 + s0 + bn;
            v4h pk;
            #pragma unroll
            for (int it = 0; it < 4; ++it) {
                float v = kp[it * TT2];
                if (!ok) v = 0.0f;
                pk[it] = (_Float16)v;
            }
            *(v4h*)&Bs[bn][bk] = pk;
        }
        // A tile (q^T) 16(t) x 32(c) per wave: 16 contiguous c per lane
        {
            const int  c_grp = imin(c0 + ah * 16, QK_C - 16);
            const bool ok    = (c0 + ah * 16 < QK_C);
            const float* qp = Qb + (size_t)c_grp * TT1 + t0 + wave * 16 + ar;
            v8h p0, p1;
            #pragma unroll
            for (int it = 0; it < 16; ++it) {
                float v = qp[it * TT1];
                if (!ok) v = 0.0f;
                if (it < 8) p0[it] = (_Float16)v;
                else        p1[it - 8] = (_Float16)v;
            }
            *(v8h*)&As[wave][ar][ah * 16]     = p0;
            *(v8h*)&As[wave][ar][ah * 16 + 8] = p1;
        }
        __syncthreads();

        v16h a, bf;
        #pragma unroll
        for (int i = 0; i < 16; ++i)
            a[i] = As[wave][m][hi * 16 + i];
        #pragma unroll
        for (int i = 0; i < 16; ++i)
            bf[i] = Bs[m][hi * 16 + i];

        acc = __builtin_amdgcn_wmma_f32_16x16x32_f16(
            false, a, false, bf, (short)0, acc, false, false);
    }

    // store: base + compile-time row-stride immediates
    {
        const int t_base = t0 + wave * 16 + 8 * hi;
        float* op = QK + ((size_t)b * TT1 + t_base) * TT2 + s0 + m;
        #pragma unroll
        for (int r = 0; r < 8; ++r)
            op[(size_t)r * TT2] = acc[r];
    }
}

// ---------------------------------------------------------------------------
// out[b*T + t] = sum_c X[b,c,t]^2   (compile-time C, T)
// ---------------------------------------------------------------------------
template <int C, int T>
__global__ __launch_bounds__(256)
void colsumsq(const float* __restrict__ X, float* __restrict__ out, int N)
{
    int i = blockIdx.x * blockDim.x + threadIdx.x;
    if (i >= N) return;
    int b = i / T, t = i - b * T;
    const float* Xb = X + (size_t)b * C * T + t;
    float s = 0.0f;
    #pragma unroll 4
    for (int c = 0; c < C; ++c) {
        float v = Xb[(size_t)c * T];
        s = fmaf(v, v, s);
    }
    out[i] = s;
}

// ---------------------------------------------------------------------------
// epilogue: per (b,t) row of length TT2=400
//   l  = -5e-4 * (q2 + k2 - 2*qk)
//   lp = log_softmax(l) + log(prior + 1e-8)       -> attn_logprob
//   attn = softmax(where(mask, -inf, lp))         -> attn
// ---------------------------------------------------------------------------
__device__ __forceinline__ float block_reduce_max(float v, float* red) {
    const int tid = threadIdx.x;
    red[tid] = v;
    __syncthreads();
    #pragma unroll
    for (int off = 64; off > 0; off >>= 1) {
        if (tid < off) red[tid] = fmaxf(red[tid], red[tid + off]);
        __syncthreads();
    }
    float r = red[0];
    __syncthreads();
    return r;
}

__device__ __forceinline__ float block_reduce_sum(float v, float* red) {
    const int tid = threadIdx.x;
    red[tid] = v;
    __syncthreads();
    #pragma unroll
    for (int off = 64; off > 0; off >>= 1) {
        if (tid < off) red[tid] += red[tid + off];
        __syncthreads();
    }
    float r = red[0];
    __syncthreads();
    return r;
}

__global__ __launch_bounds__(128)
void attn_epilogue(const float* __restrict__ QK, const float* __restrict__ q2,
                   const float* __restrict__ k2, const float* __restrict__ prior,
                   const unsigned char* __restrict__ mask,
                   float* __restrict__ attn, float* __restrict__ logprob)
{
    __shared__ float red[128];
    const int row = blockIdx.x;                 // b*TT1 + t
    const int b   = row / TT1;
    const int tid = threadIdx.x;

    const float* qkrow = QK    + (size_t)row * TT2;
    const float* prow  = prior + (size_t)row * TT2;
    float*       arow  = attn    + (size_t)row * TT2;
    float*       lrow  = logprob + (size_t)row * TT2;
    const float  q2v   = q2[row];

    float l[4];
    unsigned char mk[4];
    float mx = -FLT_MAX;
    #pragma unroll
    for (int j = 0; j < 4; ++j) {
        int s = tid + j * 128;
        float v = -FLT_MAX;
        mk[j] = 1;
        if (s < TT2) {
            v = -0.0005f * (q2v + k2[b * TT2 + s] - 2.0f * qkrow[s]);
            mk[j] = mask[b * TT2 + s];
        }
        l[j] = v;
        mx = fmaxf(mx, v);
    }
    mx = block_reduce_max(mx, red);

    float sum = 0.0f;
    #pragma unroll
    for (int j = 0; j < 4; ++j)
        if (tid + j * 128 < TT2) sum += expf(l[j] - mx);
    sum = block_reduce_sum(sum, red);
    const float logz = mx + logf(sum);

    float lp[4];
    #pragma unroll
    for (int j = 0; j < 4; ++j) {
        int s = tid + j * 128;
        lp[j] = -FLT_MAX;
        if (s < TT2) {
            lp[j] = l[j] - logz + logf(prow[s] + 1e-8f);
            lrow[s] = lp[j];
        }
    }

    float mx2 = -FLT_MAX;
    #pragma unroll
    for (int j = 0; j < 4; ++j)
        if (!mk[j]) mx2 = fmaxf(mx2, lp[j]);
    mx2 = block_reduce_max(mx2, red);

    float sum2 = 0.0f;
    #pragma unroll
    for (int j = 0; j < 4; ++j)
        if (!mk[j]) sum2 += expf(lp[j] - mx2);
    sum2 = block_reduce_sum(sum2, red);
    const float inv2 = 1.0f / sum2;

    #pragma unroll
    for (int j = 0; j < 4; ++j) {
        int s = tid + j * 128;
        if (s < TT2)
            arow[s] = mk[j] ? 0.0f : expf(lp[j] - mx2) * inv2;
    }
}

// ---------------------------------------------------------------------------
extern "C" void kernel_launch(void* const* d_in, const int* in_sizes, int n_in,
                              void* d_out, int out_size, void* d_ws, size_t ws_size,
                              hipStream_t stream) {
    const float*         queries = (const float*)d_in[0];   // [32, 80, 1600]
    const float*         keys    = (const float*)d_in[1];   // [32, 512, 400]
    const unsigned char* mask    = (const unsigned char*)d_in[2]; // [32, 400] bool
    const float*         prior   = (const float*)d_in[3];   // [32, 1600, 400]
    const float* kw1 = (const float*)d_in[4];   // [1024, 512, 3]
    const float* kb1 = (const float*)d_in[5];
    const float* kw2 = (const float*)d_in[6];   // [80, 1024, 1]
    const float* kb2 = (const float*)d_in[7];
    const float* qw1 = (const float*)d_in[8];   // [160, 80, 3]
    const float* qb1 = (const float*)d_in[9];
    const float* qw2 = (const float*)d_in[10];  // [80, 160, 1]
    const float* qb2 = (const float*)d_in[11];
    const float* qw3 = (const float*)d_in[12];  // [80, 80, 1]
    const float* qb3 = (const float*)d_in[13];

    // workspace layout (floats), liveness-based aliasing:
    //   [0,          20,480,000) : k1 (dead after keyconv2) then qk
    //   [20,480,000, 21,504,000) : k
    //   [21,504,000, 29,696,000) : q1, later qB (conv3 out)
    //   [29,696,000, 33,792,000) : qA
    //   [33,792,000, 33,843,200) : q2
    //   [33,843,200, 33,856,000) : k2
    float* ws   = (float*)d_ws;
    float* k1   = ws;
    float* qk   = ws;
    float* kbuf = ws + 20480000u;
    float* q1   = ws + 21504000u;
    float* qB   = q1;
    float* qA   = ws + 29696000u;
    float* q2   = ws + 33792000u;
    float* k2   = ws + 33843200u;

    float* attn_out = (float*)d_out;                 // [32,1,1600,400]
    float* logp_out = attn_out + 20480000u;          // [32,1,1600,400]

    // 1) k1 = relu(conv(keys, kw1, pad=1))
    conv1d_wmma<512, 1024, 400, 3, 1, true>
        <<<dim3(7, 64, BATCH), 128, 0, stream>>>(keys, kw1, kb1, k1);
    // 2) k = conv(k1, kw2)
    conv1d_wmma<1024, 80, 400, 1, 0, false>
        <<<dim3(7, 5, BATCH), 128, 0, stream>>>(k1, kw2, kb2, kbuf);
    // 3) q1 = relu(conv(queries, qw1, pad=1))
    conv1d_wmma<80, 160, 1600, 3, 1, true>
        <<<dim3(25, 10, BATCH), 128, 0, stream>>>(queries, qw1, qb1, q1);
    // 4) qA = relu(conv(q1, qw2))
    conv1d_wmma<160, 80, 1600, 1, 0, true>
        <<<dim3(25, 5, BATCH), 128, 0, stream>>>(q1, qw2, qb2, qA);
    // 5) qB = conv(qA, qw3)
    conv1d_wmma<80, 80, 1600, 1, 0, false>
        <<<dim3(25, 5, BATCH), 128, 0, stream>>>(qA, qw3, qb3, qB);
    // 6) qk[b,t,s] = sum_c qB[b,c,t]*k[b,c,s]
    qk_wmma<<<dim3(25, 25, BATCH), 128, 0, stream>>>(qB, kbuf, qk);
    // 7) squared column norms
    colsumsq<80, 1600><<<dim3((BATCH * TT1 + 255) / 256), 256, 0, stream>>>(qB, q2, BATCH * TT1);
    colsumsq<80, 400><<<dim3((BATCH * TT2 + 255) / 256), 256, 0, stream>>>(kbuf, k2, BATCH * TT2);
    // 8) softmax epilogue, writes both tuple outputs (attn, attn_logprob)
    attn_epilogue<<<dim3(BATCH * TT1), 128, 0, stream>>>(qk, q2, k2, prior, mask,
                                                         attn_out, logp_out);
}
